// RelPartialLearnableMultiHeadAttn_2911987827169
// MI455X (gfx1250) — matrix operand order repaired and checked by reading, not compiled, not run on hardware
//
#include <hip/hip_runtime.h>

// ---------------- problem constants ----------------
#define QLEN 1024
#define MLEN 1024
#define KLENN 2048
#define BSZ 4
#define NH 16
#define DH 64
#define DM 1024
#define LN_EPS 1e-5f

typedef __attribute__((ext_vector_type(16))) __bf16 v16bf;
typedef __attribute__((ext_vector_type(8)))  __bf16 v8bf;
typedef __attribute__((ext_vector_type(8)))  float  v8f;

// ---------------- WMMA helpers (CDNA5 wave32 layouts) ----------------
__device__ __forceinline__ v16bf mk16(v8bf lo, v8bf hi) {
  return __builtin_shufflevector(lo, hi, 0,1,2,3,4,5,6,7,8,9,10,11,12,13,14,15);
}

// A-matrix 16x32 bf16 fragment from row-major [16 x >=32] block, row stride ld.
// lane<16: row=lane, K = 0..7 | 16..23 ; lane>=16: row=lane-16, K = 8..15 | 24..31
__device__ __forceinline__ v16bf load_a32(const __bf16* base, int ld) {
  int lane = threadIdx.x & 31;
  int m  = lane & 15;
  int kb = (lane >> 4) << 3;
  v8bf lo = *(const v8bf*)(base + m * ld + kb);
  v8bf hi = *(const v8bf*)(base + m * ld + 16 + kb);
  return mk16(lo, hi);
}

// B-matrix 32x16 bf16 fragment. Source laid out so element (k,n) is at
// base + n*ldn + k  (k contiguous). lane<16: col=lane, K = 0..7 | 16..23 ; lane>=16: K = 8..15 | 24..31
__device__ __forceinline__ v16bf load_b32(const __bf16* base, int ldn) {
  int lane = threadIdx.x & 31;
  int n    = lane & 15;
  int koff = (lane >> 4) << 3;
  v8bf lo = *(const v8bf*)(base + n * ldn + koff);
  v8bf hi = *(const v8bf*)(base + n * ldn + 16 + koff);
  return mk16(lo, hi);
}

__device__ __forceinline__ v8f wmma_bf16(v16bf a, v16bf b, v8f c) {
  return __builtin_amdgcn_wmma_f32_16x16x32_bf16(false, a, false, b, (short)0, c, false, false);
}

// ---------------- f32 -> bf16 convert ----------------
__global__ void k_cvt(const float* __restrict__ s, __bf16* __restrict__ d, int n) {
  int i = blockIdx.x * blockDim.x + threadIdx.x;
  int stride = gridDim.x * blockDim.x;
  for (; i < n; i += stride) d[i] = (__bf16)s[i];
}

// ---------------- QKV projection: [8192,1024] x qkv_w^T -> scatter q/k/vT ----------------
// one wave = 32(M) x 64(N) tile, K = 1024, register double-buffered K loop
__global__ __launch_bounds__(128)
void k_gemm_qkv(const __bf16* __restrict__ cat, const __bf16* __restrict__ wq,
                const float* __restrict__ rwb, const float* __restrict__ rrb,
                __bf16* __restrict__ qrw, __bf16* __restrict__ qrr,
                __bf16* __restrict__ kh, __bf16* __restrict__ vT) {
  int wid = (blockIdx.x * blockDim.x + threadIdx.x) >> 5;   // [0, 256*48)
  int mt = wid / 48, ng = wid % 48;
  int m0 = mt * 32, nc0 = ng * 64;
  const __bf16* pa = cat + (size_t)m0 * DM;
  v8f acc[2][4] = {};
  v16bf A0[2], A1[2], B0[4], B1[4];

#define LD_A(buf, kc) { buf[0] = load_a32(pa + (kc), DM); buf[1] = load_a32(pa + 16*DM + (kc), DM); }
#define LD_B(buf, kc) { _Pragma("unroll") for (int t = 0; t < 4; ++t) \
                          buf[t] = load_b32(wq + (size_t)(nc0 + t*16) * DM + (kc), DM); }
#define MM(Ab, Bb)   { _Pragma("unroll") for (int t = 0; t < 4; ++t) { \
                          acc[0][t] = wmma_bf16(Ab[0], Bb[t], acc[0][t]); \
                          acc[1][t] = wmma_bf16(Ab[1], Bb[t], acc[1][t]); } }

  LD_A(A0, 0); LD_B(B0, 0);
  for (int kc = 0; kc < DM; kc += 64) {
    LD_A(A1, kc + 32); LD_B(B1, kc + 32);     // prefetch odd chunk
    MM(A0, B0);
    if (kc + 64 < DM) { LD_A(A0, kc + 64); LD_B(B0, kc + 64); }  // prefetch next even
    MM(A1, B1);
  }
#undef LD_A
#undef LD_B
#undef MM

  int lane = threadIdx.x & 31;
  int n = lane & 15, hi = lane >> 4;
#pragma unroll
  for (int mm = 0; mm < 2; ++mm)
#pragma unroll
    for (int t = 0; t < 4; ++t)
#pragma unroll
      for (int v = 0; v < 8; ++v) {
        int vrow = mm * 16 + v + 8 * hi;
        int grow = m0 + vrow;               // row = kpos*BSZ + b
        int gcol = nc0 + t * 16 + n;        // [0,3072)
        float val = acc[mm][t][v];
        int region = gcol >> 10, col = gcol & 1023;
        int h = col >> 6, dh = col & 63;
        int kpos = grow >> 2, b = grow & 3;
        if (region == 0) {
          if (kpos >= MLEN) {
            int i = kpos - MLEN;
            size_t o = ((size_t)(b * NH + h) * QLEN + i) * DH + dh;
            qrw[o] = (__bf16)(val + rwb[col]);
            qrr[o] = (__bf16)(val + rrb[col]);
          }
        } else if (region == 1) {
          kh[((size_t)(b * NH + h) * KLENN + kpos) * DH + dh] = (__bf16)val;
        } else {
          vT[((size_t)(b * NH + h) * DH + dh) * KLENN + kpos] = (__bf16)val;  // transposed!
        }
      }
}

// ---------------- r_head_k = r @ r_net_w^T, head-major [h][r][d] ----------------
// one wave = 32(M) x 64(N), double-buffered
__global__ __launch_bounds__(128)
void k_gemm_rk(const __bf16* __restrict__ rbf, const __bf16* __restrict__ wr,
               __bf16* __restrict__ rk) {
  int wid = (blockIdx.x * blockDim.x + threadIdx.x) >> 5;   // [0, 64*16)
  int mt = wid / 16, ng = wid % 16;
  int m0 = mt * 32, nc0 = ng * 64;
  const __bf16* pa = rbf + (size_t)m0 * DM;
  v8f acc[2][4] = {};
  v16bf A0[2], A1[2], B0[4], B1[4];

#define LD_A(buf, kc) { buf[0] = load_a32(pa + (kc), DM); buf[1] = load_a32(pa + 16*DM + (kc), DM); }
#define LD_B(buf, kc) { _Pragma("unroll") for (int t = 0; t < 4; ++t) \
                          buf[t] = load_b32(wr + (size_t)(nc0 + t*16) * DM + (kc), DM); }
#define MM(Ab, Bb)   { _Pragma("unroll") for (int t = 0; t < 4; ++t) { \
                          acc[0][t] = wmma_bf16(Ab[0], Bb[t], acc[0][t]); \
                          acc[1][t] = wmma_bf16(Ab[1], Bb[t], acc[1][t]); } }

  LD_A(A0, 0); LD_B(B0, 0);
  for (int kc = 0; kc < DM; kc += 64) {
    LD_A(A1, kc + 32); LD_B(B1, kc + 32);
    MM(A0, B0);
    if (kc + 64 < DM) { LD_A(A0, kc + 64); LD_B(B0, kc + 64); }
    MM(A1, B1);
  }
#undef LD_A
#undef LD_B
#undef MM

  int lane = threadIdx.x & 31;
  int n = lane & 15, hi = lane >> 4;
#pragma unroll
  for (int mm = 0; mm < 2; ++mm)
#pragma unroll
    for (int t = 0; t < 4; ++t)
#pragma unroll
      for (int v = 0; v < 8; ++v) {
        int grow = m0 + mm * 16 + v + 8 * hi;
        int gcol = nc0 + t * 16 + n;
        int h = gcol >> 6, dh = gcol & 63;
        rk[((size_t)h * KLENN + grow) * DH + dh] = (__bf16)acc[mm][t][v];
      }
}

// ---------------- fused flash attention with rel-shift ----------------
// one wave per (b, h, 16-row i-tile); 4 waves per block, private LDS regions.
__global__ __launch_bounds__(128)
void k_attn(const __bf16* __restrict__ qrw, const __bf16* __restrict__ qrr,
            const __bf16* __restrict__ kh, const __bf16* __restrict__ vT,
            const __bf16* __restrict__ rk, __bf16* __restrict__ av) {
  __shared__ float  s_strip[4][16 * 48];   // BDraw strip per wave
  __shared__ __bf16 s_p[4][16 * 32];       // P tile (C layout -> A layout) per wave
  int w    = threadIdx.x >> 5;
  int lane = threadIdx.x & 31;
  int g  = blockIdx.x * 4 + w;             // [0, 4096)
  int b  = g >> 10;
  int h  = (g >> 6) & 15;
  int i0 = (g & 63) << 4;

  const __bf16* qrw_p = qrw + ((size_t)(b * NH + h) * QLEN + i0) * DH;
  const __bf16* qrr_p = qrr + ((size_t)(b * NH + h) * QLEN + i0) * DH;
  const __bf16* kh_p  = kh  + (size_t)(b * NH + h) * KLENN * DH;
  const __bf16* vT_p  = vT  + (size_t)(b * NH + h) * DH * KLENN;
  const __bf16* rk_p  = rk  + (size_t)h * KLENN * DH;

  v16bf qa0 = load_a32(qrw_p, DH),      qa1 = load_a32(qrw_p + 32, DH);
  v16bf qb0 = load_a32(qrr_p, DH),      qb1 = load_a32(qrr_p + 32, DH);

  v8f o0 = {}, o1 = {}, o2 = {}, o3 = {};
  float mrun[8], lrun[8];
#pragma unroll
  for (int v = 0; v < 8; ++v) { mrun[v] = -1e30f; lrun[v] = 0.0f; }

  int n = lane & 15, hi = lane >> 4;
  const float scale = 0.125f;              // 1/sqrt(D_HEAD)
  float*  strip = s_strip[w];
  __bf16* pbuf  = s_p[w];

  int jend = i0 + MLEN + 16;               // last needed j = i0+15+1024
  for (int j0 = 0; j0 < jend; j0 += 32) {
    int start = j0 - i0 + 1008;            // 16-aligned base of BDraw band

    // --- prefetch next j-step's operands into WGP$/L2 (global_prefetch_b8) ---
    if (j0 + 32 < jend) {
      int nkr = j0 + 32 + lane;                       // next 32 K rows, 1 line each
      if (nkr < KLENN) __builtin_prefetch(kh_p + (size_t)nkr * DH, 0, 3);
      int nrr = start + 48 + lane;                    // next BDraw r-band
      if (nrr < KLENN) __builtin_prefetch(rk_p + (size_t)nrr * DH, 0, 3);
      // next V columns: 64 d-rows, 2 per lane; one 128B line covers 64 j's
      __builtin_prefetch(vT_p + (size_t)(lane * 2)     * KLENN + j0 + 32, 0, 3);
      __builtin_prefetch(vT_p + (size_t)(lane * 2 + 1) * KLENN + j0 + 32, 0, 3);
    }

    // --- three BDraw 16x16 r-tiles -> LDS strip ---
#pragma unroll
    for (int t = 0; t < 3; ++t) {
      int r0 = start + 16 * t;
      v8f bd = {};
      if (r0 < KLENN) {                    // uniform branch; OOB r is masked later
        bd = wmma_bf16(qb0, load_b32(rk_p + (size_t)r0 * DH, DH), bd);
        bd = wmma_bf16(qb1, load_b32(rk_p + (size_t)r0 * DH + 32, DH), bd);
      }
#pragma unroll
      for (int v = 0; v < 8; ++v)
        strip[(v + 8 * hi) * 48 + 16 * t + n] = bd[v];
    }
    // --- scores: AC (WMMA) + shifted BD read, causal mask ---
    float sv[2][8];
#pragma unroll
    for (int s = 0; s < 2; ++s) {
      v8f ac = {};
      const __bf16* kp = kh_p + (size_t)(j0 + 16 * s) * DH;
      ac = wmma_bf16(qa0, load_b32(kp, DH), ac);
      ac = wmma_bf16(qa1, load_b32(kp + 32, DH), ac);
#pragma unroll
      for (int v = 0; v < 8; ++v) {
        int vrow = v + 8 * hi;
        int i = i0 + vrow;
        int j = j0 + 16 * s + n;
        float x = (ac[v] + strip[vrow * 48 + (16 * s + n + 15 - vrow)]) * scale;
        sv[s][v] = (j > i + MLEN) ? -1e30f : x;
      }
    }
    // --- online softmax over the 16 lanes sharing each row ---
#pragma unroll
    for (int v = 0; v < 8; ++v) {
      float mx = fmaxf(sv[0][v], sv[1][v]);
      mx = fmaxf(mx, __shfl_xor(mx, 1, 32));
      mx = fmaxf(mx, __shfl_xor(mx, 2, 32));
      mx = fmaxf(mx, __shfl_xor(mx, 4, 32));
      mx = fmaxf(mx, __shfl_xor(mx, 8, 32));
      float mnew = fmaxf(mrun[v], mx);
      float corr = __expf(mrun[v] - mnew);
      float p0 = __expf(sv[0][v] - mnew);
      float p1 = __expf(sv[1][v] - mnew);
      float ps = p0 + p1;
      ps += __shfl_xor(ps, 1, 32);
      ps += __shfl_xor(ps, 2, 32);
      ps += __shfl_xor(ps, 4, 32);
      ps += __shfl_xor(ps, 8, 32);
      lrun[v] = lrun[v] * corr + ps;
      mrun[v] = mnew;
      o0[v] *= corr; o1[v] *= corr; o2[v] *= corr; o3[v] *= corr;
      int vrow = v + 8 * hi;
      pbuf[vrow * 32 + n]      = (__bf16)p0;
      pbuf[vrow * 32 + 16 + n] = (__bf16)p1;
    }
    // --- P (A frag via LDS) x V^T (contiguous B frags) ---
    {
      int m  = lane & 15;
      int kb = (lane >> 4) << 3;
      v8bf lo = *(const v8bf*)(pbuf + m * 32 + kb);
      v8bf hp = *(const v8bf*)(pbuf + m * 32 + 16 + kb);
      v16bf pa = mk16(lo, hp);
      o0 = wmma_bf16(pa, load_b32(vT_p + (size_t)0 * 16 * KLENN + j0, KLENN), o0);
      o1 = wmma_bf16(pa, load_b32(vT_p + (size_t)1 * 16 * KLENN + j0, KLENN), o1);
      o2 = wmma_bf16(pa, load_b32(vT_p + (size_t)2 * 16 * KLENN + j0, KLENN), o2);
      o3 = wmma_bf16(pa, load_b32(vT_p + (size_t)3 * 16 * KLENN + j0, KLENN), o3);
    }
  }
  // --- normalize, write attn_vec rows (i*BSZ+b, h*64+d) as bf16 ---
#pragma unroll
  for (int v = 0; v < 8; ++v) {
    int vrow = v + 8 * hi;
    float inv = 1.0f / lrun[v];
    int grow = (i0 + vrow) * BSZ + b;
    size_t base = (size_t)grow * DM + h * DH;
    av[base +  0 + n] = (__bf16)(o0[v] * inv);
    av[base + 16 + n] = (__bf16)(o1[v] * inv);
    av[base + 32 + n] = (__bf16)(o2[v] * inv);
    av[base + 48 + n] = (__bf16)(o3[v] * inv);
  }
}

// ---------------- output projection + residual + LayerNorm ----------------
// block = 16 rows x 1024 cols: 16 waves, wave w -> cols [w*64, w*64+64)
__global__ __launch_bounds__(512)
void k_out_ln(const __bf16* __restrict__ av, const __bf16* __restrict__ ow,
              const float* __restrict__ wres, const float* __restrict__ gamma,
              const float* __restrict__ beta, float* __restrict__ out) {
  __shared__ float s_x[16 * 1024];
  int w = threadIdx.x >> 5, lane = threadIdx.x & 31;
  int r0 = blockIdx.x * 16;
  const __bf16* pa = av + (size_t)r0 * DM;
  v8f acc[4] = {};
  v16bf A0c, A1c, B0[4], B1[4];

#define LD_B(buf, kc) { _Pragma("unroll") for (int t = 0; t < 4; ++t) \
                          buf[t] = load_b32(ow + (size_t)(w*64 + t*16) * DM + (kc), DM); }
#define MM(Ab, Bb)   { _Pragma("unroll") for (int t = 0; t < 4; ++t) \
                          acc[t] = wmma_bf16(Ab, Bb[t], acc[t]); }

  A0c = load_a32(pa, DM); LD_B(B0, 0);
  for (int kc = 0; kc < DM; kc += 64) {
    A1c = load_a32(pa + kc + 32, DM); LD_B(B1, kc + 32);
    MM(A0c, B0);
    if (kc + 64 < DM) { A0c = load_a32(pa + kc + 64, DM); LD_B(B0, kc + 64); }
    MM(A1c, B1);
  }
#undef LD_B
#undef MM

  int n = lane & 15, hi = lane >> 4;
#pragma unroll
  for (int t = 0; t < 4; ++t)
#pragma unroll
    for (int v = 0; v < 8; ++v)
      s_x[(v + 8 * hi) * 1024 + w * 64 + t * 16 + n] = acc[t][v];
  __syncthreads();

  // wave w normalizes row w: 32 lanes x 32 elements each
  int grow = r0 + w;
  const float* wr = wres + (size_t)grow * DM;
  const float* xr = s_x + w * 1024;
  float s = 0.0f, s2 = 0.0f;
#pragma unroll 4
  for (int k = 0; k < 32; ++k) {
    float x = wr[lane * 32 + k] + xr[lane * 32 + k];
    s += x; s2 += x * x;
  }
#pragma unroll
  for (int m = 1; m <= 16; m <<= 1) {
    s  += __shfl_xor(s,  m, 32);
    s2 += __shfl_xor(s2, m, 32);
  }
  float mu  = s  * (1.0f / DM);
  float var = s2 * (1.0f / DM) - mu * mu;
  float rs  = rsqrtf(var + LN_EPS);
  float* orow = out + (size_t)grow * DM;
#pragma unroll 4
  for (int k = 0; k < 32; ++k) {
    int col = lane * 32 + k;
    float x = wr[col] + xr[col];
    orow[col] = (x - mu) * rs * gamma[col] + beta[col];
  }
}

// ---------------- host orchestration ----------------
extern "C" void kernel_launch(void* const* d_in, const int* in_sizes, int n_in,
                              void* d_out, int out_size, void* d_ws, size_t ws_size,
                              hipStream_t stream) {
  (void)in_sizes; (void)n_in; (void)out_size; (void)ws_size;
  const float* w_in   = (const float*)d_in[0];   // [1024,4,1024]
  const float* r_in   = (const float*)d_in[1];   // [2048,1024]
  const float* mems   = (const float*)d_in[2];   // [1024,4,1024]
  /* d_in[3] = attn_mask: analytic (j > i+MLEN), not read */
  const float* qkv_w  = (const float*)d_in[4];   // [3072,1024]
  const float* rnet_w = (const float*)d_in[5];   // [1024,1024]
  const float* o_w    = (const float*)d_in[6];   // [1024,1024]
  const float* rrb    = (const float*)d_in[7];   // r_r_bias [16,64]
  const float* rwb    = (const float*)d_in[8];   // r_w_bias [16,64]
  const float* gamma  = (const float*)d_in[9];
  const float* beta   = (const float*)d_in[10];
  float* out = (float*)d_out;

  char* ws = (char*)d_ws;
  size_t off = 0;
  __bf16* cat_bf  = (__bf16*)(ws + off); off += (size_t)KLENN * BSZ * DM * 2;   // 16 MB
  __bf16* qkvw_bf = (__bf16*)(ws + off); off += (size_t)3 * DM * DM * 2;        // 6 MB
  __bf16* rnet_bf = (__bf16*)(ws + off); off += (size_t)DM * DM * 2;            // 2 MB
  __bf16* ow_bf   = (__bf16*)(ws + off); off += (size_t)DM * DM * 2;            // 2 MB
  __bf16* r_bf    = (__bf16*)(ws + off); off += (size_t)KLENN * DM * 2;         // 4 MB
  __bf16* qrw     = (__bf16*)(ws + off); off += (size_t)BSZ * NH * QLEN * DH * 2;  // 8 MB
  __bf16* qrr     = (__bf16*)(ws + off); off += (size_t)BSZ * NH * QLEN * DH * 2;  // 8 MB
  __bf16* kh      = (__bf16*)(ws + off); off += (size_t)BSZ * NH * KLENN * DH * 2; // 16 MB
  __bf16* vT      = (__bf16*)(ws + off); off += (size_t)BSZ * NH * DH * KLENN * 2; // 16 MB
  __bf16* rk      = (__bf16*)(ws + off); off += (size_t)NH * KLENN * DH * 2;       // 4 MB
  __bf16* av      = (__bf16*)(ws + off); off += (size_t)QLEN * BSZ * DM * 2;       // 8 MB

  const int NW = QLEN * BSZ * DM;       // 4,194,304
  // cat = [mems ; w]
  k_cvt<<<4096, 256, 0, stream>>>(mems,  cat_bf, NW);
  k_cvt<<<4096, 256, 0, stream>>>(w_in,  cat_bf + (size_t)NW, NW);
  k_cvt<<<4096, 256, 0, stream>>>(qkv_w, qkvw_bf, 3 * DM * DM);
  k_cvt<<<2048, 256, 0, stream>>>(rnet_w, rnet_bf, DM * DM);
  k_cvt<<<2048, 256, 0, stream>>>(o_w,   ow_bf,   DM * DM);
  k_cvt<<<4096, 256, 0, stream>>>(r_in,  r_bf,    KLENN * DM);

  // QKV: 256 M-tiles x 48 N-groups = 12288 waves
  k_gemm_qkv<<<3072, 128, 0, stream>>>(cat_bf, qkvw_bf, rwb, rrb, qrw, qrr, kh, vT);
  // r_head_k: 64 x 16 = 1024 waves
  k_gemm_rk<<<256, 128, 0, stream>>>(r_bf, rnet_bf, rk);
  // attention: 4*16*64 = 4096 waves
  k_attn<<<1024, 128, 0, stream>>>(qrw, qrr, kh, vT, rk, av);
  // output projection + LN: 4096 rows / 16 per block
  k_out_ln<<<256, 512, 0, stream>>>(av, ow_bf, w_in, gamma, beta, out);
}
